// Attn_head_adj_9543417332153
// MI455X (gfx1250) — matrix operand order repaired and checked by reading, not compiled, not run on hardware
//
#include <hip/hip_runtime.h>
#include <hip/hip_bf16.h>
#include <math.h>

typedef __attribute__((ext_vector_type(16))) __bf16 v16bf;
typedef __attribute__((ext_vector_type(8)))  float  v8f;
typedef __attribute__((ext_vector_type(8)))  int    v8i;
typedef __attribute__((ext_vector_type(8)))  float  v8fv;

#define B_ 4
#define N_ 4096
#define C_ 64
#define F_ 64

// ---------------------------------------------------------------------------
// Kernel 1: seq_fts = x@W1 + b1; store TRANSPOSED bf16 copy seqbT[b][f][n]
// (so kernel 3's WMMA B-fragments are contiguous 32B runs), plus
// f1 = seq_fts@a1 + ba1, f2 = seq_fts@a2 + ba2.
// Block handles 64 rows; W1 + x-rows in LDS; LDS tile transpose on output.
// ---------------------------------------------------------------------------
__global__ __launch_bounds__(256)
void gat_seqfts_kernel(const float* __restrict__ x,  const float* __restrict__ W1,
                       const float* __restrict__ b1, const float* __restrict__ a1,
                       const float* __restrict__ ba1,const float* __restrict__ a2,
                       const float* __restrict__ ba2,
                       __bf16* __restrict__ seqbT, float* __restrict__ f1,
                       float* __restrict__ f2)
{
    __shared__ float  Ws[64][65];
    __shared__ float  Xs[64][65];
    __shared__ __bf16 Ts[64][72];              // [f][row-in-block], padded
    const int tid  = threadIdx.x;
    const int row0 = blockIdx.x * 64;          // flat (b*N + n) row base
    const int b    = row0 / N_;
    const int n0   = row0 % N_;

    for (int t = tid; t < 64 * 64; t += 256) {
        Ws[t >> 6][t & 63] = W1[t];
        Xs[t >> 6][t & 63] = x[(size_t)(row0 + (t >> 6)) * C_ + (t & 63)];
    }
    __syncthreads();

    const int r  = tid >> 2;          // row within block (0..63)
    const int c0 = (tid & 3) * 16;    // 16-wide column group

    float acc[16];
    #pragma unroll
    for (int c = 0; c < 16; ++c) acc[c] = b1[c0 + c];

    #pragma unroll 4
    for (int k = 0; k < 64; ++k) {
        const float xv = Xs[r][k];
        #pragma unroll
        for (int c = 0; c < 16; ++c) acc[c] = fmaf(xv, Ws[k][c0 + c], acc[c]);
    }

    float p1 = 0.f, p2 = 0.f;
    #pragma unroll
    for (int c = 0; c < 16; ++c) {
        Ts[c0 + c][r] = (__bf16)acc[c];        // transpose via LDS
        p1 = fmaf(acc[c], a1[c0 + c], p1);
        p2 = fmaf(acc[c], a2[c0 + c], p2);
    }
    // reduce the 4 column groups (consecutive lanes) of one row
    p1 += __shfl_xor(p1, 1, 32); p1 += __shfl_xor(p1, 2, 32);
    p2 += __shfl_xor(p2, 1, 32); p2 += __shfl_xor(p2, 2, 32);
    if ((tid & 3) == 0) {
        f1[row0 + r] = p1 + ba1[0];
        f2[row0 + r] = p2 + ba2[0];
    }
    __syncthreads();
    // coalesced transposed write-out: 64 f-rows x 64 contiguous columns
    for (int t = tid; t < 64 * 64; t += 256) {
        const int f  = t >> 6;
        const int rr = t & 63;
        seqbT[((size_t)(b * F_ + f)) * N_ + n0 + rr] = Ts[f][rr];
    }
}

// ---------------------------------------------------------------------------
// Kernel 2: per-column-j online softmax stats over i (softmax axis=1).
//   cls[b,j] = colmax + log(colsum)   => pass 3 uses exp(v - cls).
// One thread per column; adj streamed row-wise, fully coalesced, read once.
// ---------------------------------------------------------------------------
__global__ __launch_bounds__(256)
void gat_colstats_kernel(const int* __restrict__ adj, const float* __restrict__ f1,
                         const float* __restrict__ f2, float* __restrict__ cls)
{
    const int b   = blockIdx.y;
    const int j   = blockIdx.x * 256 + threadIdx.x;
    const int bN  = b * N_;
    const float f1j = f1[bN + j];
    const int* __restrict__ acol = adj + (size_t)bN * N_ + j;

    float m = -INFINITY, s = 0.f;
    #pragma unroll 4
    for (int i = 0; i < N_; ++i) {
        if ((i & 7) == 0 && i + 16 < N_)
            __builtin_prefetch(&acol[(size_t)(i + 16) * N_], 0, 0);
        const int av = acol[(size_t)i * N_];
        float v = f1j + f2[bN + i];
        v = fmaxf(v, 0.01f * v);                 // leaky_relu(0.01)
        if (av > 0) {
            if (v > m) { s = s * __expf(m - v) + 1.f; m = v; }
            else       { s += __expf(v - m); }
        }
    }
    cls[bN + j] = (s > 0.f) ? (m + __logf(s)) : 0.f;
}

// ---------------------------------------------------------------------------
// Kernel 3: ret = elu(P @ V), P[i,j] = adj ? exp(lrelu(f1[j]+f2[i]) - cls[j]) : 0
// LDS-free / barrier-free: every WMMA operand is loaded straight from global
// memory in fragment order.
//   A (16x32 P tile, 16-bit layout): lane m reads adj/f1/cls at
//     j0 + {kbA..kbA+7, kbA+8..kbA+15}  -> two contiguous 32B runs (b128s).
//   B (32x16 V tile): lane n reads seqbT[b][nt*16+n][j0+16*half .. +15]
//     -> 16 contiguous bf16 = one v16bf load (2 x b128).  seqbT is 2MB,
//     L2-resident, so all 1024 row-tiles re-read it from L2, while adj
//     streams from HBM exactly once (fully consumed cache lines).
// ---------------------------------------------------------------------------
__global__ __launch_bounds__(256)
void gat_attn_wmma_kernel(const int* __restrict__ adj, const __bf16* __restrict__ seqbT,
                          const float* __restrict__ f1, const float* __restrict__ f2,
                          const float* __restrict__ cls, float* __restrict__ out)
{
    const int tid  = threadIdx.x;
    const int wave = tid >> 5;
    const int lane = tid & 31;
    const int b    = blockIdx.y;
    const int bN   = b * N_;
    const int i0   = blockIdx.x * 128 + wave * 16;   // 8 waves x 16 rows

    const int m    = lane & 15;            // A row / B col / D col
    const int hi   = lane >> 4;            // wave half
    const int kbA  = hi * 16;              // A fragment K base (contiguous, matches B)
    const int kbB  = hi * 16;              // B fragment K base

    const float f2i = f2[bN + i0 + m];
    const int* __restrict__ arow = adj + ((size_t)bN + i0 + m) * N_ + kbA;

    // the four B-fragment source rows (f = nt*16 + m), contiguous in j
    const __bf16* __restrict__ vr0 = seqbT + ((size_t)(b * F_ +  0 + m)) * N_ + kbB;
    const __bf16* __restrict__ vr1 = seqbT + ((size_t)(b * F_ + 16 + m)) * N_ + kbB;
    const __bf16* __restrict__ vr2 = seqbT + ((size_t)(b * F_ + 32 + m)) * N_ + kbB;
    const __bf16* __restrict__ vr3 = seqbT + ((size_t)(b * F_ + 48 + m)) * N_ + kbB;

    v8f acc0 = {}, acc1 = {}, acc2 = {}, acc3 = {};

    for (int j0 = 0; j0 < N_; j0 += 32) {
        // --- A operands: adjacency + f1 + cls runs (each 2 x 32B, b128s) ---
        const v8i  alo = *(const v8i *)(arow + j0);          // cols kbA..kbA+7
        const v8i  ahi = *(const v8i *)(arow + j0 + 8);      // cols kbA+8..+15
        const v8fv flo = *(const v8fv*)(f1  + bN + j0 + kbA);
        const v8fv fhi = *(const v8fv*)(f1  + bN + j0 + kbA + 8);
        const v8fv clo = *(const v8fv*)(cls + bN + j0 + kbA);
        const v8fv chi = *(const v8fv*)(cls + bN + j0 + kbA + 8);
        if (j0 + 32 < N_) __builtin_prefetch(arow + j0 + 32, 0, 0);

        // --- build A fragment: P[i0+m, j0+k] in bf16 ---
        v16bf afrag;
        #pragma unroll
        for (int e = 0; e < 16; ++e) {
            const int   av = (e < 8) ? alo[e] : ahi[e - 8];
            const float fj = (e < 8) ? flo[e] : fhi[e - 8];
            const float cj = (e < 8) ? clo[e] : chi[e - 8];
            float v = fj + f2i;
            v = fmaxf(v, 0.01f * v);                 // leaky_relu(0.01)
            const float p = (av > 0) ? __expf(v - cj) : 0.f;
            afrag[e] = (__bf16)p;
        }

        // --- B fragments: contiguous v16bf loads, then 4 WMMAs ---
        const v16bf b0 = *(const v16bf*)(vr0 + j0);
        const v16bf b1 = *(const v16bf*)(vr1 + j0);
        const v16bf b2 = *(const v16bf*)(vr2 + j0);
        const v16bf b3 = *(const v16bf*)(vr3 + j0);
        acc0 = __builtin_amdgcn_wmma_f32_16x16x32_bf16(false, afrag, false, b0,
                                                       (short)0, acc0, false, false);
        acc1 = __builtin_amdgcn_wmma_f32_16x16x32_bf16(false, afrag, false, b1,
                                                       (short)0, acc1, false, false);
        acc2 = __builtin_amdgcn_wmma_f32_16x16x32_bf16(false, afrag, false, b2,
                                                       (short)0, acc2, false, false);
        acc3 = __builtin_amdgcn_wmma_f32_16x16x32_bf16(false, afrag, false, b3,
                                                       (short)0, acc3, false, false);
    }

    // ELU + store per ISA C/D layout: VGPR r -> row (r + 8*hi), col = lane&15
    const int mrow = hi * 8;
    #pragma unroll
    for (int r = 0; r < 8; ++r) {
        const size_t o = ((size_t)bN + i0 + mrow + r) * F_ + m;
        float v;
        v = acc0[r]; out[o +  0] = (v > 0.f) ? v : (__expf(v) - 1.f);
        v = acc1[r]; out[o + 16] = (v > 0.f) ? v : (__expf(v) - 1.f);
        v = acc2[r]; out[o + 32] = (v > 0.f) ? v : (__expf(v) - 1.f);
        v = acc3[r]; out[o + 48] = (v > 0.f) ? v : (__expf(v) - 1.f);
    }
}

// ---------------------------------------------------------------------------
extern "C" void kernel_launch(void* const* d_in, const int* in_sizes, int n_in,
                              void* d_out, int out_size, void* d_ws, size_t ws_size,
                              hipStream_t stream)
{
    const float* x   = (const float*)d_in[0];
    const int*   adj = (const int*)  d_in[1];
    const float* W1  = (const float*)d_in[2];
    const float* b1  = (const float*)d_in[3];
    const float* a1  = (const float*)d_in[4];
    const float* ba1 = (const float*)d_in[5];
    const float* a2  = (const float*)d_in[6];
    const float* ba2 = (const float*)d_in[7];
    float* out = (float*)d_out;

    char*   ws    = (char*)d_ws;
    __bf16* seqbT = (__bf16*)ws;                               // B*F*N bf16 (2 MB)
    float*  f1    = (float*)(ws + (size_t)B_ * F_ * N_ * 2);   // B*N f32
    float*  f2    = f1 + B_ * N_;
    float*  cls   = f2 + B_ * N_;

    gat_seqfts_kernel<<<dim3((B_ * N_) / 64), dim3(256), 0, stream>>>(
        x, W1, b1, a1, ba1, a2, ba2, seqbT, f1, f2);
    gat_colstats_kernel<<<dim3(N_ / 256, B_), dim3(256), 0, stream>>>(
        adj, f1, f2, cls);
    gat_attn_wmma_kernel<<<dim3(N_ / 128, B_), dim3(256), 0, stream>>>(
        adj, seqbT, f1, f2, cls, out);
}
